// YOLOV11_73555609911850
// MI455X (gfx1250) — matrix operand order repaired and checked by reading, not compiled, not run on hardware
//
#include <hip/hip_runtime.h>
#include <stdint.h>

typedef __attribute__((ext_vector_type(16))) _Float16 v16h;
typedef __attribute__((ext_vector_type(8)))  float    v8f;
typedef __attribute__((ext_vector_type(4)))  unsigned int v4u;
typedef __attribute__((ext_vector_type(4)))  int v4i;
typedef __attribute__((ext_vector_type(8)))  int v8i;

#define BATCH 32
#define RMAX  16
#define ATOT  8400
#define NGT   20
#define NIGN  2
#define BA    (BATCH*ATOT)        /* 268800  */
#define BNA   (BATCH*NGT*ATOT)    /* 5376000 */
#define NBLK_BA 1050              /* BA/256  */
#define EPSF  1e-9f

#ifndef __has_builtin
#define __has_builtin(x) 0
#endif
#if defined(__AMDGCN__) && __has_builtin(__builtin_amdgcn_tensor_load_to_lds)
#define HAVE_TDM 1
#endif

// ---- compile-probe diagnostics (read these in stderr next round) ----
#define PROBE_STR2(x) #x
#define PROBE_STR(x) PROBE_STR2(x)
#if defined(__AMDGCN__)
#pragma message("PROBE device pass: clang_major=" PROBE_STR(__clang_major__))
#  if defined(HAVE_TDM)
#    warning "PROBE device pass: HAVE_TDM=1 (tensor_load_to_lds builtin available)"
#  else
#    warning "PROBE device pass: HAVE_TDM=0 (tensor_load_to_lds builtin NOT available)"
#  endif
#endif

// ---------------- helpers ----------------
__device__ __forceinline__ void anchor_of(int a, float& gx, float& gy, float& st){
  int i, w;
  if (a < 6400)      { i = a;        w = 80; st = 8.f;  }
  else if (a < 8000) { i = a - 6400; w = 40; st = 16.f; }
  else               { i = a - 8000; w = 20; st = 32.f; }
  gx = (float)(i % w) + 0.5f;
  gy = (float)(i / w) + 0.5f;
}

__device__ __forceinline__ const float* feat_base(const float* f0, const float* f1,
                                                  const float* f2, int b, int a,
                                                  int& idx, int& hw){
  if (a < 6400)      { hw = 6400; idx = a;        return f0 + (size_t)b*65*6400; }
  else if (a < 8000) { hw = 1600; idx = a - 6400; return f1 + (size_t)b*65*1600; }
  else               { hw = 400;  idx = a - 8000; return f2 + (size_t)b*65*400;  }
}

// CIoU exactly matching the reference (box1 vs box2 ordering matters)
__device__ __forceinline__ float ciou_f(float b1x1,float b1y1,float b1x2,float b1y2,
                                        float b2x1,float b2y1,float b2x2,float b2y2){
  const float eps = 1e-7f;
  float w1 = b1x2-b1x1, h1 = b1y2-b1y1 + eps;
  float w2 = b2x2-b2x1, h2 = b2y2-b2y1 + eps;
  float iw = fminf(b1x2,b2x2) - fmaxf(b1x1,b2x1); iw = fmaxf(iw, 0.f);
  float ih = fminf(b1y2,b2y2) - fmaxf(b1y1,b2y1); ih = fmaxf(ih, 0.f);
  float inter = iw*ih;
  float uni = w1*h1 + w2*h2 - inter + eps;
  float iou = inter/uni;
  float cw = fmaxf(b1x2,b2x2) - fminf(b1x1,b2x1);
  float ch = fmaxf(b1y2,b2y2) - fminf(b1y1,b2y1);
  float c2 = cw*cw + ch*ch + eps;
  float dx = (b2x1+b2x2-b1x1-b1x2), dy = (b2y1+b2y2-b1y1-b1y2);
  float rho2 = (dx*dx + dy*dy)*0.25f;
  float dv = atanf(w2/h2) - atanf(w1/h1);
  float v = 0.4052847345693511f * dv * dv;     /* 4/pi^2 */
  float alpha = v / (v - iou + (1.0f + eps));
  return iou - (rho2/c2 + v*alpha);
}

// ---------------- kernels ----------------
__global__ void k_zero(float* p, long n){
  long t = (long)blockIdx.x*blockDim.x + threadIdx.x;
  if (t < n) p[t] = 0.f;
}

__global__ void k_prep(const float* gtx, const float* ignx,
                       float* gtb, float* gtm, float* ignb){
  int t = blockIdx.x*blockDim.x + threadIdx.x;
  if (t < BATCH*NGT){
    float cx = gtx[t*4+0]*640.f, cy = gtx[t*4+1]*640.f;
    float w  = gtx[t*4+2]*640.f, h  = gtx[t*4+3]*640.f;
    float x1 = cx - 0.5f*w, y1 = cy - 0.5f*h, x2 = cx + 0.5f*w, y2 = cy + 0.5f*h;
    gtb[t*4+0]=x1; gtb[t*4+1]=y1; gtb[t*4+2]=x2; gtb[t*4+3]=y2;
    gtm[t] = ((x1+y1+x2+y2) > 0.f) ? 1.f : 0.f;
  }
  if (t < BATCH*NIGN){
    float cx = ignx[t*4+0]*640.f, cy = ignx[t*4+1]*640.f;
    float w  = ignx[t*4+2]*640.f, h  = ignx[t*4+3]*640.f;
    ignb[t*4+0]=cx-0.5f*w; ignb[t*4+1]=cy-0.5f*h;
    ignb[t*4+2]=cx+0.5f*w; ignb[t*4+3]=cy+0.5f*h;
  }
}

// DFL decode: dist[b,a,side] = softmax(logits16) . [0..15] via v_wmma_f32_16x16x32_f16
// Wave layout: row m = lane%16 = local_anchor*4 + side; lane<16 owns bins 0..7,
// lane>=16 owns bins 8..15 (ISA 16-bit A 16x32 layout; K 16..31 zero-padded).
// B matrix: proj[k]=k broadcast across all 16 columns (lanes 0..15 hold K0..7,
// lanes 16..31 hold K8..15; K16..31 = 0). Every column of D is the numerator.
__global__ void k_decode(const float* f0, const float* f1, const float* f2, float* distb){
  int lane = threadIdx.x & 31;
  int wid  = blockIdx.x*(blockDim.x >> 5) + (threadIdx.x >> 5);
  int half = lane >> 4;
  int m    = lane & 15;
  int ancl = m >> 2, side = m & 3;

  v16h bv;
  #pragma unroll
  for (int j = 0; j < 16; ++j) bv[j] = (j < 8) ? (_Float16)(j + 8*half) : (_Float16)0;

  // writer mapping: lanes 0..7 write rows 0..7; lanes 16..23 write rows 8..15.
  bool writer = (lane < 8) | (lane >= 16 && lane < 24);
  int  wrow   = (lane < 8) ? lane : (lane - 8);     // valid only for writers
  int  dsrc   = (lane >= 16 && lane < 24) ? (lane - 8) : lane;
  int  csel   = lane & 7;                           // static-select index

  #pragma unroll
  for (int t = 0; t < 8; ++t){
    int gbase = wid*32 + t*4;            // 4 anchors per WMMA
    int g = gbase + ancl;                // exact coverage: grid*32 == BA
    int b = g / ATOT, a = g % ATOT;
    int idx, hw; const float* base = feat_base(f0, f1, f2, b, a, idx, hw);

    float l8[8]; float mx = -INFINITY;
    #pragma unroll
    for (int j = 0; j < 8; ++j){
      int bin = j + 8*half;
      l8[j] = base[(size_t)(side*16 + bin)*hw + idx];
      mx = fmaxf(mx, l8[j]);
    }
    mx = fmaxf(mx, __shfl_xor(mx, 16, 32));
    float s = 0.f; v16h av;
    #pragma unroll
    for (int j = 0; j < 16; ++j) av[j] = (_Float16)0;
    #pragma unroll
    for (int j = 0; j < 8; ++j){ float e = __expf(l8[j]-mx); s += e; av[j] = (_Float16)e; }
    float stot = s + __shfl_xor(s, 16, 32);

    v8f c;
    #pragma unroll
    for (int j = 0; j < 8; ++j) c[j] = 0.f;
    c = __builtin_amdgcn_wmma_f32_16x16x32_f16(false, av, false, bv,
                                               (short)0, c, false, false);

    // branch-free static-index select of this lane's numerator component
    float num = 0.f;
    #pragma unroll
    for (int v = 0; v < 8; ++v) num = (csel == v) ? c[v] : num;
    float den = __shfl(stot, dsrc, 32);   // row's softmax denominator

    if (writer){
      int ga = gbase + (wrow >> 2);
      int wb = ga / ATOT, wa = ga % ATOT;
      distb[((size_t)wb*ATOT + wa)*4 + (wrow & 3)] = num / den;
    }
  }
}

__global__ void k_score(const float* f0, const float* f1, const float* f2,
                        float* raw, float* pd){
  int t = blockIdx.x*blockDim.x + threadIdx.x; if (t >= BA) return;
  int b = t / ATOT, a = t % ATOT;
  int idx, hw; const float* base = feat_base(f0, f1, f2, b, a, idx, hw);
  float r = base[(size_t)64*hw + idx];
  raw[t] = r;
  pd[t]  = 1.f/(1.f + __expf(-r));
}

// pairwise overlaps & align over (b, n, a)
__global__ void k_pair(const float* distb, const float* pd, const float* gtb,
                       const float* gtm, float* ovl, float* aln){
  size_t t = (size_t)blockIdx.x*blockDim.x + threadIdx.x;
  if (t >= (size_t)BNA) return;
  int a = (int)(t % ATOT); size_t bn = t / ATOT;
  int n = (int)(bn % NGT), b = (int)(bn / NGT);
  float gx, gy, st; anchor_of(a, gx, gy, st);
  float ax = gx*st, ay = gy*st;
  const float* g = &gtb[(b*NGT + n)*4];
  float mi = fminf(fminf(ax - g[0], ay - g[1]), fminf(g[2] - ax, g[3] - ay));
  float maskv = (mi > EPSF) ? gtm[b*NGT + n] : 0.f;
  const float* d = &distb[((size_t)b*ATOT + a)*4];
  float px1 = (gx - d[0])*st, py1 = (gy - d[1])*st;
  float px2 = (gx + d[2])*st, py2 = (gy + d[3])*st;
  float ci = ciou_f(g[0], g[1], g[2], g[3], px1, py1, px2, py2);
  float o  = fmaxf(ci, 0.f) * maskv;
  float cls = pd[(size_t)b*ATOT + a];                 // NC==1 -> class 0
  float o2 = o*o;
  ovl[t] = o;
  aln[t] = sqrtf(fmaxf(cls, 0.f)) * (o2*o2*o2) * maskv;
}

// top-10 per (b,n): stage 8400-float align row into LDS with the TDM, then
// 10 block-wide argmax passes marking mask_topk into mp (zero-initialized).
__global__ void k_topk(const float* aln, const float* gtm, float* mp){
  __shared__ float vals[ATOT];
  __shared__ float rmax[256];
  __shared__ int   ridx[256];
  int bn = blockIdx.x;
  const float* row = aln + (size_t)bn*ATOT;

#if defined(HAVE_TDM)
  if (threadIdx.x < 32){
    unsigned long long ga = (unsigned long long)(uintptr_t)row;
    unsigned int lds = (unsigned int)(uintptr_t)(&vals[0]);
    v4u g0; g0[0] = 1u;                              // count=1
    g0[1] = lds;                                     // lds_addr [63:32]
    g0[2] = (unsigned int)ga;                        // global_addr lo
    g0[3] = ((unsigned int)(ga >> 32) & 0x1FFFFFFu) | (2u << 30);  // addr hi | type=2
    v8i g1;
    g1[0] = (int)(2u << 16);                         // data_size=4B, no mask/pad
    g1[1] = (int)((8400u & 0xFFFFu) << 16);          // tensor_dim0 lo16 @ [63:48]
    g1[2] = (int)((8400u >> 16) | (1u << 16));       // tensor_dim0 hi | tensor_dim1=1
    g1[3] = (int)(8400u << 16);                      // tile_dim0 @ [127:112]
    g1[4] = 1;                                       // tile_dim1=1, tile_dim2=0
    g1[5] = 8400;                                    // tensor_dim0_stride lo32
    g1[6] = (int)((8400u & 0xFFFFu) << 16);          // stride hi=0 | dim1_stride lo16
    g1[7] = 0;
    v4i g2 = {0,0,0,0};
    v4i g3 = {0,0,0,0};
#if __clang_major__ >= 23
    v8i g4 = {0,0,0,0,0,0,0,0};
    __builtin_amdgcn_tensor_load_to_lds(g0, g1, g2, g3, g4, 0);
#else
    __builtin_amdgcn_tensor_load_to_lds(g0, g1, g2, g3, 0);
#endif
    __builtin_amdgcn_s_wait_tensorcnt(0);
  }
#else
  for (int i = threadIdx.x; i < ATOT; i += blockDim.x) vals[i] = row[i];
#endif
  __syncthreads();
  if (gtm[bn] <= 0.f) return;              // invalid gt: no marks (uniform exit)

  for (int it = 0; it < 10; ++it){
    float bm = -INFINITY; int bi = 0;
    for (int i = threadIdx.x; i < ATOT; i += blockDim.x){
      float v = vals[i];
      if (v > bm){ bm = v; bi = i; }
    }
    rmax[threadIdx.x] = bm; ridx[threadIdx.x] = bi;
    __syncthreads();
    for (int ofs = 128; ofs > 0; ofs >>= 1){
      if (threadIdx.x < ofs){
        float vo = rmax[threadIdx.x + ofs]; int io = ridx[threadIdx.x + ofs];
        float vm = rmax[threadIdx.x];       int im = ridx[threadIdx.x];
        if (vo > vm || (vo == vm && io < im)){ rmax[threadIdx.x] = vo; ridx[threadIdx.x] = io; }
      }
      __syncthreads();
    }
    if (threadIdx.x == 0){
      int w = ridx[0];
      mp[(size_t)bn*ATOT + w] = 1.0f;
      vals[w] = -INFINITY;
    }
    __syncthreads();
  }
}

// per (b,a): mask_pos = topk * in_gts * gt_mask; multi-assign tie-break; fg/tgt
__global__ void k_resolve(const float* ovl, float* mp, const float* gtb,
                          const float* gtm, int* fg, int* tgt){
  int t = blockIdx.x*blockDim.x + threadIdx.x; if (t >= BA) return;
  int b = t / ATOT, a = t % ATOT;
  float gx, gy, st; anchor_of(a, gx, gy, st);
  float ax = gx*st, ay = gy*st;
  float m[NGT]; float fgs = 0.f;
  for (int n = 0; n < NGT; ++n){
    const float* g = &gtb[(b*NGT + n)*4];
    float mi = fminf(fminf(ax - g[0], ay - g[1]), fminf(g[2] - ax, g[3] - ay));
    float inm = (mi > EPSF) ? 1.f : 0.f;
    float v = mp[((size_t)(b*NGT + n))*ATOT + a] * inm * gtm[b*NGT + n];
    m[n] = v; fgs += v;
  }
  if (fgs > 1.f){
    float bo = -INFINITY; int bi = 0;
    for (int n = 0; n < NGT; ++n){
      float o = ovl[((size_t)(b*NGT + n))*ATOT + a];
      if (o > bo){ bo = o; bi = n; }
    }
    for (int n = 0; n < NGT; ++n) m[n] = (n == bi) ? 1.f : 0.f;
    fgs = 1.f;
  }
  float best = m[0]; int ti = 0;
  for (int n = 1; n < NGT; ++n) if (m[n] > best){ best = m[n]; ti = n; }
  fg[t]  = (fgs > 0.f) ? 1 : 0;
  tgt[t] = ti;
  for (int n = 0; n < NGT; ++n) mp[((size_t)(b*NGT + n))*ATOT + a] = m[n];
}

__global__ void k_posmax(const float* aln, const float* ovl, const float* mp,
                         float* pa, float* po){
  __shared__ float ra[256], ro[256];
  int bn = blockIdx.x;
  float ma = 0.f, mo = 0.f;                 // all candidates are >= 0
  for (int i = threadIdx.x; i < ATOT; i += 256){
    size_t id = (size_t)bn*ATOT + i;
    float msk = mp[id];
    ma = fmaxf(ma, aln[id]*msk);
    mo = fmaxf(mo, ovl[id]*msk);
  }
  ra[threadIdx.x] = ma; ro[threadIdx.x] = mo;
  __syncthreads();
  for (int ofs = 128; ofs > 0; ofs >>= 1){
    if (threadIdx.x < ofs){
      ra[threadIdx.x] = fmaxf(ra[threadIdx.x], ra[threadIdx.x + ofs]);
      ro[threadIdx.x] = fmaxf(ro[threadIdx.x], ro[threadIdx.x + ofs]);
    }
    __syncthreads();
  }
  if (threadIdx.x == 0){ pa[bn] = ra[0]; po[bn] = ro[0]; }
}

__global__ void k_norm(const float* aln, const float* mp, const float* pa,
                       const float* po, const int* fg, float* tsb){
  int t = blockIdx.x*blockDim.x + threadIdx.x; if (t >= BA) return;
  int b = t / ATOT, a = t % ATOT;
  float nm = 0.f;
  for (int n = 0; n < NGT; ++n){
    int bn = b*NGT + n;
    size_t id = (size_t)bn*ATOT + a;
    nm = fmaxf(nm, aln[id]*mp[id]*po[bn]/(pa[bn] + EPSF));
  }
  tsb[t] = fg[t] ? nm : 0.f;                // target_scores sum (NC==1)
}

__global__ void k_ign(const float* distb, const float* ignb, int* fg){
  int t = blockIdx.x*blockDim.x + threadIdx.x; if (t >= BA) return;
  int b = t / ATOT, a = t % ATOT;
  float gx, gy, st; anchor_of(a, gx, gy, st);
  const float* d = &distb[(size_t)t*4];
  float px1 = (gx - d[0])*st, py1 = (gy - d[1])*st;
  float px2 = (gx + d[2])*st, py2 = (gy + d[3])*st;
  float cx = 0.5f*(px1 + px2), cy = 0.5f*(py1 + py2);
  bool ig = false;
  for (int k = 0; k < NIGN; ++k){
    const float* ib = &ignb[(b*NIGN + k)*4];
    ig = ig || (cx >= ib[0] && cx < ib[2] && cy >= ib[1] && cy < ib[3]);
  }
  if (ig) fg[t] = 0;
}

// fused BCE + CIoU + DFL, deterministic per-block partials (no atomics)
__global__ void k_loss(const float* f0, const float* f1, const float* f2,
                       const float* distb, const float* raw, const float* tsb,
                       const int* fg, const int* tgt, const float* gtb,
                       float* part){
  __shared__ float red[4][256];
  int t = blockIdx.x*blockDim.x + threadIdx.x;
  float s_ts = 0.f, s_bce = 0.f, s_iou = 0.f, s_dfl = 0.f;
  if (t < BA){
    int b = t / ATOT, a = t % ATOT;
    float fgf = fg[t] ? 1.f : 0.f;
    float ts  = tsb[t];
    s_ts = ts * fgf;
    float r = raw[t];
    float bce = fmaxf(r, 0.f) - r*ts + log1pf(__expf(-fabsf(r)));
    s_bce = bce * fgf;
    float weight = ts * fgf;

    float gx, gy, st; anchor_of(a, gx, gy, st);
    const float* d = &distb[(size_t)t*4];
    float px1 = gx - d[0], py1 = gy - d[1], px2 = gx + d[2], py2 = gy + d[3];
    const float* g = &gtb[(b*NGT + tgt[t])*4];
    float tb0 = g[0]/st, tb1 = g[1]/st, tb2 = g[2]/st, tb3 = g[3]/st;
    float iou = ciou_f(px1, py1, px2, py2, tb0, tb1, tb2, tb3);
    s_iou = (1.f - iou)*weight;

    int idx, hw; const float* base = feat_base(f0, f1, f2, b, a, idx, hw);
    float dsum = 0.f;
    #pragma unroll
    for (int fs = 0; fs < 4; ++fs){
      float td = (fs == 0) ? (gx - tb0) : (fs == 1) ? (gy - tb1)
               : (fs == 2) ? (tb2 - gx) : (tb3 - gy);
      td = fminf(fmaxf(td, 0.f), (float)RMAX - 1.f - 0.01f);
      int tl = (int)floorf(td); int tr = tl + 1;
      float wl = (float)tr - td; float wr = 1.f - wl;
      float l[16]; float mx = -INFINITY;
      #pragma unroll
      for (int rr = 0; rr < 16; ++rr){
        l[rr] = base[(size_t)(fs*16 + rr)*hw + idx];
        mx = fmaxf(mx, l[rr]);
      }
      float se = 0.f;
      #pragma unroll
      for (int rr = 0; rr < 16; ++rr) se += __expf(l[rr] - mx);
      float lse = mx + __logf(se);
      dsum += (lse - l[tl])*wl + (lse - l[tr])*wr;
    }
    s_dfl = dsum*0.25f*weight;
  }
  red[0][threadIdx.x] = s_ts; red[1][threadIdx.x] = s_bce;
  red[2][threadIdx.x] = s_iou; red[3][threadIdx.x] = s_dfl;
  __syncthreads();
  for (int ofs = 128; ofs > 0; ofs >>= 1){
    if (threadIdx.x < ofs)
      for (int j = 0; j < 4; ++j)
        red[j][threadIdx.x] += red[j][threadIdx.x + ofs];
    __syncthreads();
  }
  if (threadIdx.x == 0)
    for (int j = 0; j < 4; ++j) part[blockIdx.x*4 + j] = red[j][0];
}

__global__ void k_fin(const float* part, int nblk, float* out){
  if (blockIdx.x == 0 && threadIdx.x == 0){
    float s0 = 0.f, s1 = 0.f, s2 = 0.f, s3 = 0.f;
    for (int i = 0; i < nblk; ++i){
      s0 += part[i*4+0]; s1 += part[i*4+1]; s2 += part[i*4+2]; s3 += part[i*4+3];
    }
    float tss = fmaxf(s0, 1.0f);
    out[0] = 7.5f * s2 / tss * (float)BATCH;   // box
    out[1] = 0.5f * s1 / tss * (float)BATCH;   // cls
    out[2] = 1.5f * s3 / tss * (float)BATCH;   // dfl
  }
}

// ---------------- workspace layout (floats) ----------------
static constexpr size_t OFF_DIST = 0;
static constexpr size_t OFF_RAW  = OFF_DIST + (size_t)BA*4;
static constexpr size_t OFF_PD   = OFF_RAW  + (size_t)BA;
static constexpr size_t OFF_GTB  = OFF_PD   + (size_t)BA;
static constexpr size_t OFF_GTM  = OFF_GTB  + (size_t)BATCH*NGT*4;
static constexpr size_t OFF_IGN  = OFF_GTM  + (size_t)BATCH*NGT;
static constexpr size_t OFF_OVL  = OFF_IGN  + (size_t)BATCH*NIGN*4;
static constexpr size_t OFF_ALN  = OFF_OVL  + (size_t)BNA;
static constexpr size_t OFF_MP   = OFF_ALN  + (size_t)BNA;
static constexpr size_t OFF_FG   = OFF_MP   + (size_t)BNA;
static constexpr size_t OFF_TGT  = OFF_FG   + (size_t)BA;
static constexpr size_t OFF_PA   = OFF_TGT  + (size_t)BA;
static constexpr size_t OFF_PO   = OFF_PA   + (size_t)BATCH*NGT;
static constexpr size_t OFF_TS   = OFF_PO   + (size_t)BATCH*NGT;
static constexpr size_t OFF_PART = OFF_TS   + (size_t)BA;

extern "C" void kernel_launch(void* const* d_in, const int* in_sizes, int n_in,
                              void* d_out, int out_size, void* d_ws, size_t ws_size,
                              hipStream_t stream){
  (void)in_sizes; (void)n_in; (void)out_size; (void)ws_size;
  const float* f0   = (const float*)d_in[0];
  const float* f1   = (const float*)d_in[1];
  const float* f2   = (const float*)d_in[2];
  const float* gtx  = (const float*)d_in[3];
  const int*   gtc  = (const int*)d_in[4]; (void)gtc;  // NC==1 -> labels all 0
  const float* ignx = (const float*)d_in[5];
  float* ws   = (float*)d_ws;
  float* out  = (float*)d_out;

  float* dist = ws + OFF_DIST;
  float* raw  = ws + OFF_RAW;
  float* pd   = ws + OFF_PD;
  float* gtb  = ws + OFF_GTB;
  float* gtm  = ws + OFF_GTM;
  float* ignb = ws + OFF_IGN;
  float* ovl  = ws + OFF_OVL;
  float* aln  = ws + OFF_ALN;
  float* mp   = ws + OFF_MP;
  int*   fg   = (int*)(ws + OFF_FG);
  int*   tgt  = (int*)(ws + OFF_TGT);
  float* pa   = ws + OFF_PA;
  float* po   = ws + OFF_PO;
  float* tsb  = ws + OFF_TS;
  float* part = ws + OFF_PART;

  k_zero   <<<(BNA + 255)/256, 256, 0, stream>>>(mp, (long)BNA);
  k_prep   <<<3, 256, 0, stream>>>(gtx, ignx, gtb, gtm, ignb);
  k_decode <<<NBLK_BA, 256, 0, stream>>>(f0, f1, f2, dist);
  k_score  <<<NBLK_BA, 256, 0, stream>>>(f0, f1, f2, raw, pd);
  k_pair   <<<BNA/256, 256, 0, stream>>>(dist, pd, gtb, gtm, ovl, aln);
  k_topk   <<<BATCH*NGT, 256, 0, stream>>>(aln, gtm, mp);
  k_resolve<<<NBLK_BA, 256, 0, stream>>>(ovl, mp, gtb, gtm, fg, tgt);
  k_posmax <<<BATCH*NGT, 256, 0, stream>>>(aln, ovl, mp, pa, po);
  k_norm   <<<NBLK_BA, 256, 0, stream>>>(aln, mp, pa, po, fg, tsb);
  k_ign    <<<NBLK_BA, 256, 0, stream>>>(dist, ignb, fg);
  k_loss   <<<NBLK_BA, 256, 0, stream>>>(f0, f1, f2, dist, raw, tsb, fg, tgt, gtb, part);
  k_fin    <<<1, 32, 0, stream>>>(part, NBLK_BA, out);
}
